// GLaRE_86251533238765
// MI455X (gfx1250) — matrix operand order repaired and checked by reading, not compiled, not run on hardware
//
#include <hip/hip_runtime.h>

// ---------------------------------------------------------------------------
// GLaRE GNN forward for MI455X (gfx1250, wave32, WMMA 16x16x32 f16 -> f32 acc)
//
// tgt = repeat(arange(N), DEG), DEG == 16  ->  one wave per node:
//   A (16 x K f16) edge messages in LDS -> WMMA MLP -> in-register segment-max
// Intermediates kept in f16; A built with 16B v8h packets (ds_store_b128) and
// packed f16 subtracts; target-feature block broadcast cooperatively.
// ---------------------------------------------------------------------------

#define N_NODES 100000
#define DEG     16
#define HDIM    64
#define NPB     8            // nodes (waves) per 256-thread block

typedef __attribute__((ext_vector_type(16))) _Float16 v16h;
typedef __attribute__((ext_vector_type(8)))  _Float16 v8h;
typedef __attribute__((ext_vector_type(8)))  float    v8f;

union V16 { v16h v; v8h h[2]; };

// ---------------------------------------------------------------------------
// h0 prep: f16 [N][32]; cols 0..15 = x, 16..18 = pos, 19..31 = 0
// ---------------------------------------------------------------------------
__global__ void prep_h0_kernel(const float* __restrict__ x,
                               const float* __restrict__ pos,
                               _Float16* __restrict__ h0) {
  int i = blockIdx.x * blockDim.x + threadIdx.x;
  if (i >= N_NODES * 32) return;
  int c = i & 31, n = i >> 5;
  float v = 0.f;
  if (c < 16)      v = x[(size_t)n * 16 + c];
  else if (c < 19) v = pos[(size_t)n * 3 + (c - 16)];
  h0[i] = (_Float16)v;
}

// ---------------------------------------------------------------------------
// Weight swizzle: W (Kact x 64 f32, row major) -> f16 B-matrix lane layout.
// Per (kc, nt, lane L): 16 f16 = W[k][n], n = nt*16 + (L&15),
//   kA = kc*32 + (j>>3)*16 + (L>>4)*8 + (j&7).  If split: A K-space is [ht(0..18) pad | diff
//   (32..50) pad], i.e. source k = kA (kA<19) or kA-13 (32<=kA<51).
// ---------------------------------------------------------------------------
__global__ void swizzle_weights_kernel(const float* __restrict__ W,
                                       _Float16* __restrict__ out,
                                       int Kact, int nkc, int split) {
  int total = nkc * 4 * 32 * 16;
  for (int i = blockIdx.x * blockDim.x + threadIdx.x; i < total;
       i += gridDim.x * blockDim.x) {
    int j    =  i        & 15;
    int lane = (i >> 4)  & 31;
    int nt   = (i >> 9)  & 3;
    int kc   =  i >> 11;
    int n  = nt * 16 + (lane & 15);
    int kA = kc * 32 + (j >> 3) * 16 + (lane >> 4) * 8 + (j & 7);
    int k; int valid;
    if (split) {
      if (kA < 19)                    { k = kA;      valid = 1; }
      else if (kA >= 32 && kA < 51)   { k = kA - 13; valid = 1; }
      else                            { k = 0;       valid = 0; }
    } else {
      k = kA; valid = (k < Kact);
    }
    out[i] = valid ? (_Float16)W[k * 64 + n] : (_Float16)0.f;
  }
}

__global__ void zero_kernel(float* __restrict__ p, int n) {
  int i = blockIdx.x * blockDim.x + threadIdx.x;
  if (i < n) p[i] = 0.f;
}

// ---------------------------------------------------------------------------
// One wave: A (16 x KPAD f16 in LDS) -> layer1(ReLU) -> M (16x64 f16 LDS) ->
// layer2 -> +b2 -> max over 16 edge rows -> outer ReLU -> out[64] f16.
// Wave-private LDS; DS ops are in-order so no barrier needed.
// ---------------------------------------------------------------------------
template <int KPAD, int NKC1>
__device__ __forceinline__ void mlp16_maxout(const _Float16* __restrict__ A,
                                             _Float16* __restrict__ M,
                                             const _Float16* __restrict__ w1s,
                                             const float* __restrict__ b1,
                                             const _Float16* __restrict__ w2s,
                                             const float* __restrict__ b2,
                                             _Float16* __restrict__ out,
                                             int lane) {
  const int m  = lane & 15;
  const int kh = lane >> 4;

  // ---- layer 1: (16 x KPAD) @ (KPAD x 64) ----
#pragma unroll
  for (int nt = 0; nt < 4; ++nt) {
    v8f acc = {};
#pragma unroll
    for (int kc = 0; kc < NKC1; ++kc) {
      V16 a;
      const _Float16* ap = A + m * KPAD + kc * 32 + kh * 8;
      a.h[0] = *(const v8h*)(ap);        // K = kc*32 + kh*8 + [0..7]
      a.h[1] = *(const v8h*)(ap + 16);   // K = kc*32 + 16 + kh*8 + [0..7]
      v16h b = *(const v16h*)(w1s + ((kc * 4 + nt) * 32 + lane) * 16);
      acc = __builtin_amdgcn_wmma_f32_16x16x32_f16(
          false, a.v, false, b, (short)0, acc, false, false);
    }
    const int col = nt * 16 + m;
    const float bb = b1[col];
#pragma unroll
    for (int r = 0; r < 8; ++r) {
      float v = acc[r] + bb;
      v = v > 0.f ? v : 0.f;                       // inner ReLU
      M[(r + 8 * kh) * 64 + col] = (_Float16)v;    // C layout -> row major LDS
    }
  }

  // ---- layer 2: (16 x 64) @ (64 x 64), then segment-max over 16 rows ----
#pragma unroll
  for (int nt = 0; nt < 4; ++nt) {
    v8f acc = {};
#pragma unroll
    for (int kc = 0; kc < 2; ++kc) {
      V16 a;
      const _Float16* ap = M + m * 64 + kc * 32 + kh * 8;
      a.h[0] = *(const v8h*)(ap);
      a.h[1] = *(const v8h*)(ap + 16);
      v16h b = *(const v16h*)(w2s + ((kc * 4 + nt) * 32 + lane) * 16);
      acc = __builtin_amdgcn_wmma_f32_16x16x32_f16(
          false, a.v, false, b, (short)0, acc, false, false);
    }
    const int col = nt * 16 + m;
    float mx = acc[0];
#pragma unroll
    for (int r = 1; r < 8; ++r) mx = fmaxf(mx, acc[r]);  // rows kh*8..kh*8+7
    mx += b2[col];                                       // bias commutes w/max
    mx = fmaxf(mx, __shfl_xor(mx, 16, 32));              // join the two halves
    mx = fmaxf(mx, 0.f);  // 16 in-edges per node -> finite; outer ReLU
    if (kh == 0) out[col] = (_Float16)mx;
  }
}

// ---------------------------------------------------------------------------
// conv1: input h0 f16 [N][32] (pre-padded). A is 16x64:
//   chunk0 (cols 0..31)  = h_t  (identical for all 16 rows -> broadcast fill)
//   chunk1 (cols 32..63) = h_s - h_t (packed f16 subtract; pads diff to 0)
// ---------------------------------------------------------------------------
__global__ __launch_bounds__(256) void conv1_kernel(
    const _Float16* __restrict__ h0, const int* __restrict__ esrc,
    const _Float16* __restrict__ w1s, const float* __restrict__ b1,
    const _Float16* __restrict__ w2s, const float* __restrict__ b2,
    _Float16* __restrict__ hout) {
  __shared__ __align__(16) _Float16 lds[NPB * (16 * 64 + 16 * 64)];
  const int wave = threadIdx.x >> 5;
  const int lane = threadIdx.x & 31;
  const int node = blockIdx.x * NPB + wave;   // grid == N/NPB exactly
  _Float16* A = lds + wave * (16 * 64 + 16 * 64);
  _Float16* M = A + 16 * 64;
  v8h* Av = (v8h*)A;                          // row stride: 8 v8h

  const v8h* hn = (const v8h*)(h0 + (size_t)node * 32);   // 4 chunks
  {  // broadcast h_t into chunk0 of all 16 rows: slot = row*4 + ch
    const int ch  = lane & 3;
    const v8h tv  = hn[ch];
    const int r0  = lane >> 2;                // 0..7
    Av[r0 * 8 + ch]       = tv;
    Av[(r0 + 8) * 8 + ch] = tv;
  }
  {  // diff block: 2 lanes per edge, 2 chunks each
    const int e  = lane & 15;
    const int hh = lane >> 4;
    const int s  = esrc[node * DEG + e];
    const v8h* hs = (const v8h*)(h0 + (size_t)s * 32);
#pragma unroll
    for (int ch = hh * 2; ch < hh * 2 + 2; ++ch)
      Av[e * 8 + 4 + ch] = hs[ch] - hn[ch];   // v_pk_sub_f16; pads -> 0
  }
  mlp16_maxout<64, 2>(A, M, w1s, b1, w2s, b2, hout + (size_t)node * HDIM, lane);
}

// ---------------------------------------------------------------------------
// conv2: input h1 f16 [N][64]. A is 16x128: chunks 0..7 = h_t, 8..15 = diff.
// ---------------------------------------------------------------------------
__global__ __launch_bounds__(256) void conv2_kernel(
    const _Float16* __restrict__ hin, const int* __restrict__ esrc,
    const _Float16* __restrict__ w1s, const float* __restrict__ b1,
    const _Float16* __restrict__ w2s, const float* __restrict__ b2,
    _Float16* __restrict__ hout) {
  __shared__ __align__(16) _Float16 lds[NPB * (16 * 128 + 16 * 64)];
  const int wave = threadIdx.x >> 5;
  const int lane = threadIdx.x & 31;
  const int node = blockIdx.x * NPB + wave;
  _Float16* A = lds + wave * (16 * 128 + 16 * 64);
  _Float16* M = A + 16 * 128;
  v8h* Av = (v8h*)A;                          // row stride: 16 v8h

  const v8h* hn = (const v8h*)(hin + (size_t)node * HDIM);  // 8 chunks
  {  // broadcast h_t: slot = row*8 + ch; lane -> ch = lane&7, rows lane>>3+4k
    const int ch = lane & 7;
    const v8h tv = hn[ch];
    const int r0 = lane >> 3;                 // 0..3
#pragma unroll
    for (int k = 0; k < 4; ++k) Av[(r0 + 4 * k) * 16 + ch] = tv;
  }
  {  // diff block: 2 lanes per edge, 4 chunks each
    const int e  = lane & 15;
    const int hh = lane >> 4;
    const int s  = esrc[node * DEG + e];
    const v8h* hs = (const v8h*)(hin + (size_t)s * HDIM);
#pragma unroll
    for (int ch = hh * 4; ch < hh * 4 + 4; ++ch)
      Av[e * 16 + 8 + ch] = hs[ch] - hn[ch];
  }
  mlp16_maxout<128, 4>(A, M, w1s, b1, w2s, b2, hout + (size_t)node * HDIM, lane);
}

// ---------------------------------------------------------------------------
// Quotient pooling: per-region sums/counts of h2 (f16 in, f32 accumulate).
// ---------------------------------------------------------------------------
__global__ __launch_bounds__(256) void qpool_kernel(
    const _Float16* __restrict__ h2, const int* __restrict__ labels,
    float* __restrict__ qsum, float* __restrict__ qcnt) {
  __shared__ float lsum[8 * 64];
  __shared__ float lcnt[8];
  const int t = threadIdx.x;
  for (int i = t; i < 512; i += 256) lsum[i] = 0.f;
  if (t < 8) lcnt[t] = 0.f;
  __syncthreads();
  const int col = t & 63;
  const int sub = t >> 6;  // 0..3
  const int base = blockIdx.x * 1024;
  for (int i = base + sub; i < base + 1024 && i < N_NODES; i += 4) {
    int l = labels[i];
    atomicAdd(&lsum[l * 64 + col], (float)h2[(size_t)i * HDIM + col]);
    if (col == 0) atomicAdd(&lcnt[l], 1.f);
  }
  __syncthreads();
  for (int i = t; i < 512; i += 256) atomicAdd(&qsum[i], lsum[i]);
  if (t < 8) atomicAdd(&qcnt[t], lcnt[t]);
}

// ---------------------------------------------------------------------------
// Quotient graph (R=8, 32 edges): conv3, conv4, global_add_pool, linear.
// ---------------------------------------------------------------------------
__global__ __launch_bounds__(256) void quotient_kernel(
    const float* __restrict__ qsum, const float* __restrict__ qcnt,
    const int* __restrict__ qedge,
    const float* __restrict__ W3a, const float* __restrict__ b3a,
    const float* __restrict__ W3b, const float* __restrict__ b3b,
    const float* __restrict__ W4a, const float* __restrict__ b4a,
    const float* __restrict__ W4b, const float* __restrict__ b4b,
    const float* __restrict__ linW, const float* __restrict__ linb,
    float* __restrict__ out) {
  __shared__ float qx[8 * 64];
  __shared__ float m1[32 * 64];
  __shared__ float m2[32 * 64];
  __shared__ float emb[64];
  __shared__ int se[32], te[32];
  const int t = threadIdx.x;
  if (t < 32) { se[t] = qedge[t]; te[t] = qedge[32 + t]; }
  for (int i = t; i < 512; i += 256) {
    float c = qcnt[i >> 6];
    qx[i] = (c > 0.f) ? qsum[i] / fmaxf(c, 1.f) : 0.f;
  }
  __syncthreads();

  for (int conv = 0; conv < 2; ++conv) {
    const float* W1 = conv ? W4a : W3a; const float* B1 = conv ? b4a : b3a;
    const float* W2 = conv ? W4b : W3b; const float* B2 = conv ? b4b : b3b;
    for (int o = t; o < 32 * 64; o += 256) {   // layer 1
      int e = o >> 6, c = o & 63;
      int s = se[e], g = te[e];
      float sum = B1[c];
      for (int k = 0; k < 64; ++k) {
        sum += qx[g * 64 + k] * W1[k * 64 + c];
        sum += (qx[s * 64 + k] - qx[g * 64 + k]) * W1[(64 + k) * 64 + c];
      }
      m1[o] = fmaxf(sum, 0.f);
    }
    __syncthreads();
    for (int o = t; o < 32 * 64; o += 256) {   // layer 2
      int e = o >> 6, c = o & 63;
      float sum = B2[c];
      for (int k = 0; k < 64; ++k) sum += m1[e * 64 + k] * W2[k * 64 + c];
      m2[o] = sum;
    }
    __syncthreads();
    for (int o = t; o < 512; o += 256) {       // segment-max (+where/ReLU)
      int r = o >> 6, c = o & 63;
      float mx = -3.0e38f;
      for (int e = 0; e < 32; ++e)
        if (te[e] == r) mx = fmaxf(mx, m2[e * 64 + c]);
      qx[o] = fmaxf(mx, 0.f);
    }
    __syncthreads();
  }

  if (t < 64) {
    float s = 0.f;
    for (int r = 0; r < 8; ++r) s += qx[r * 64 + t];
    emb[t] = s;
  }
  __syncthreads();
  if (t < 8) {
    float s = linb[t];
    for (int k = 0; k < 64; ++k) s += emb[k] * linW[k * 8 + t];
    out[t] = s;
  }
}

// ---------------------------------------------------------------------------
extern "C" void kernel_launch(void* const* d_in, const int* in_sizes, int n_in,
                              void* d_out, int out_size, void* d_ws,
                              size_t ws_size, hipStream_t stream) {
  const float* x      = (const float*)d_in[0];
  const float* pos    = (const float*)d_in[1];
  const int*   eidx   = (const int*)d_in[2];   // [2][E]; row0 = src
  const int*   labels = (const int*)d_in[3];
  const int*   qedge  = (const int*)d_in[4];   // [2][32]
  const float* c1W1 = (const float*)d_in[5];  const float* c1b1 = (const float*)d_in[6];
  const float* c1W2 = (const float*)d_in[7];  const float* c1b2 = (const float*)d_in[8];
  const float* c2W1 = (const float*)d_in[9];  const float* c2b1 = (const float*)d_in[10];
  const float* c2W2 = (const float*)d_in[11]; const float* c2b2 = (const float*)d_in[12];
  const float* c3W1 = (const float*)d_in[13]; const float* c3b1 = (const float*)d_in[14];
  const float* c3W2 = (const float*)d_in[15]; const float* c3b2 = (const float*)d_in[16];
  const float* c4W1 = (const float*)d_in[17]; const float* c4b1 = (const float*)d_in[18];
  const float* c4W2 = (const float*)d_in[19]; const float* c4b2 = (const float*)d_in[20];
  const float* linW = (const float*)d_in[21]; const float* linb = (const float*)d_in[22];

  // workspace layout (all chunks 16B/32B aligned)
  _Float16* h0 = (_Float16*)d_ws;                   // N*32 f16
  _Float16* h1 = h0 + (size_t)N_NODES * 32;         // N*64 f16
  _Float16* h2 = h1 + (size_t)N_NODES * 64;         // N*64 f16
  float* qsum  = (float*)(h2 + (size_t)N_NODES * 64);  // 8*64
  float* qcnt  = qsum + 8 * 64;                     // 8
  _Float16* w1c1s = (_Float16*)(qcnt + 8);          // 2 kc * 4 nt * 512 half
  _Float16* w2c1s = w1c1s + 2 * 4 * 512;
  _Float16* w1c2s = w2c1s + 2 * 4 * 512;            // 4 kc
  _Float16* w2c2s = w1c2s + 4 * 4 * 512;

  const int* esrc = eidx;  // row 0 of edge_index

  prep_h0_kernel<<<(N_NODES * 32 + 255) / 256, 256, 0, stream>>>(x, pos, h0);
  swizzle_weights_kernel<<<8, 256, 0, stream>>>(c1W1, w1c1s, 38, 2, 1);
  swizzle_weights_kernel<<<8, 256, 0, stream>>>(c1W2, w2c1s, 64, 2, 0);
  swizzle_weights_kernel<<<16, 256, 0, stream>>>(c2W1, w1c2s, 128, 4, 0);
  swizzle_weights_kernel<<<8, 256, 0, stream>>>(c2W2, w2c2s, 64, 2, 0);
  zero_kernel<<<3, 256, 0, stream>>>(qsum, 8 * 64 + 8);

  conv1_kernel<<<N_NODES / NPB, 256, 0, stream>>>(h0, esrc, w1c1s, c1b1,
                                                  w2c1s, c1b2, h1);
  conv2_kernel<<<N_NODES / NPB, 256, 0, stream>>>(h1, esrc, w1c2s, c2b1,
                                                  w2c2s, c2b2, h2);
  qpool_kernel<<<(N_NODES + 1023) / 1024, 256, 0, stream>>>(h2, labels, qsum,
                                                            qcnt);
  quotient_kernel<<<1, 256, 0, stream>>>(qsum, qcnt, qedge, c3W1, c3b1, c3W2,
                                         c3b2, c4W1, c4b1, c4W2, c4b2, linW,
                                         linb, (float*)d_out);
}
